// GATLayer_58368605553169
// MI455X (gfx1250) — matrix operand order repaired and checked by reading, not compiled, not run on hardware
//
#include <hip/hip_runtime.h>
#include <hip/hip_bf16.h>

// ---------------------------------------------------------------------------
// GAT layer for MI455X (gfx1250, wave32).
//  feat = x@W via v_wmma_f32_16x16x32_bf16 (bf16 in, f32 accum),
//  2-stage software pipeline (double-buffered A/B fragments, unroll-pinned);
//  then edge softmax + scatter (L2-resident, atomic-bound).
// ---------------------------------------------------------------------------

#define N_NODES 50000
#define N_EDGES 800000
#define IN_CH   1024
#define HEADS   8
#define OUT_CH  32
#define NHC     (HEADS * OUT_CH)      // 256
#define M_TILES (N_NODES / 16)        // 3125
#define K_TILES (IN_CH / 32)          // 32
#define N_TILES (NHC / 16)            // 16
#define KT_STRIDE (N_TILES * 32 * 16) // ushorts per K-step of Wpack = 8192

typedef __attribute__((ext_vector_type(16))) __bf16 v16bf;
typedef __attribute__((ext_vector_type(8)))  float  v8f;
typedef __attribute__((ext_vector_type(4)))  float  v4f;

static __device__ __forceinline__ unsigned short f2bf_bits(float f) {
    // round-to-nearest-even fp32 -> bf16
    unsigned u = __float_as_uint(f);
    unsigned r = u + 0x7FFFu + ((u >> 16) & 1u);
    return (unsigned short)(r >> 16);
}

// A-fragment holder: 16 bf16
union AFrag {
    v16bf v;
    unsigned short s[16];
};

// Convert four v4f chunks (16 floats) into an A fragment (RNE).
static __device__ __forceinline__ void cvt_afrag(AFrag& A, v4f a0, v4f a1, v4f a2, v4f a3) {
#pragma unroll
    for (int j = 0; j < 4; ++j) {
        A.s[j]      = f2bf_bits(a0[j]);
        A.s[4 + j]  = f2bf_bits(a1[j]);
        A.s[8 + j]  = f2bf_bits(a2[j]);
        A.s[12 + j] = f2bf_bits(a3[j]);
    }
}

struct B8 { v16bf b[8]; };

static __device__ __forceinline__ void loadB(B8& B, const unsigned short* wb) {
#pragma unroll
    for (int g = 0; g < 8; ++g)
        B.b[g] = *reinterpret_cast<const v16bf*>(wb + g * 512);
}

static __device__ __forceinline__ void mma8(v8f acc[8], const AFrag& A, const B8& B) {
#pragma unroll
    for (int g = 0; g < 8; ++g)
        acc[g] = __builtin_amdgcn_wmma_f32_16x16x32_bf16(
            false, A.v, false, B.b[g], (short)0, acc[g], false, false);
}

// ---------------------------------------------------------------------------
// K0: pack W [1024,256] fp32 -> bf16 B-fragments, contiguous k order (A uses the same).
// B 32x16 (KxN), 16-bit: lane L -> col = L&15, K-half = L>>4;
// fragment element j (j=0..15) = W[k0 + 16*(L>>4) + j][col].
// ---------------------------------------------------------------------------
__global__ __launch_bounds__(256) void gat_pack_w(const float* __restrict__ W,
                                                  unsigned short* __restrict__ Wpack) {
    int tid  = blockIdx.x * 256 + threadIdx.x;       // 0 .. 16383
    int lane = tid & 31;
    int nt   = (tid >> 5) & 15;
    int kt   = tid >> 9;
    int col  = nt * 16 + (lane & 15);
    int k0   = kt * 32 + (lane >> 4) * 16;

    union { unsigned short s[16]; uint4 q[2]; } u;
#pragma unroll
    for (int j = 0; j < 16; ++j)
        u.s[j] = f2bf_bits(W[(size_t)(k0 + j) * NHC + col]);

    uint4* dst = reinterpret_cast<uint4*>(Wpack + (size_t)tid * 16);
    dst[0] = u.q[0];
    dst[1] = u.q[1];
}

// ---------------------------------------------------------------------------
// K1: feat[n,256] = x[n,1024] @ W, bf16 WMMA with f32 accumulation.
// Block = 256 thr = 8 waves; wave handles one 16-row M-tile x 128 cols
// (8 accumulator tiles). Hand 2-stage pipeline, unroll pinned to 1 so
// register pressure stays < 256 VGPRs (no vgpr_msb, no phi copies).
// ---------------------------------------------------------------------------
__global__ __launch_bounds__(256) void gat_gemm(const float* __restrict__ x,
                                                const unsigned short* __restrict__ Wpack,
                                                float* __restrict__ feat) {
    const int tid   = threadIdx.x;
    const int w     = tid >> 5;
    const int lane  = tid & 31;
    const int mtile = blockIdx.x * 4 + (w >> 1);
    if (mtile >= M_TILES) return;                    // uniform per wave
    const int ngrp  = w & 1;                         // 0 -> cols 0..127, 1 -> 128..255

    const int r = mtile * 16 + (lane & 15);          // A row for this lane
    const int h = lane >> 4;                         // K-half selector
    const float* xrow = x + (size_t)r * IN_CH + h * 16;
    const unsigned short* wbase = Wpack + ((size_t)(ngrp * 8) * 32 + lane) * 16;

    v8f acc[8] = {};                                 // 8 x (16x16 f32) tiles

    // ---- prologue: stage kt=0 into buffer 0 ----
    AFrag A0, A1;
    B8    B0, B1;
    {
        v4f a0 = *(const v4f*)(xrow);
        v4f a1 = *(const v4f*)(xrow + 4);
        v4f a2 = *(const v4f*)(xrow + 8);
        v4f a3 = *(const v4f*)(xrow + 12);
        cvt_afrag(A0, a0, a1, a2, a3);
    }
    loadB(B0, wbase);

    // ---- steady state: kt = 0,2,...,28 (last double-step peeled) ----
#pragma unroll 1
    for (int kt = 0; kt < K_TILES - 2; kt += 2) {
        // issue loads for step kt+1 into buffer 1 (no wait yet)
        loadB(B1, wbase + (size_t)(kt + 1) * KT_STRIDE);
        const float* p1 = xrow + (kt + 1) * 32;
        v4f n0 = *(const v4f*)(p1);
        v4f n1 = *(const v4f*)(p1 + 4);
        v4f n2 = *(const v4f*)(p1 + 8);
        v4f n3 = *(const v4f*)(p1 + 12);

        mma8(acc, A0, B0);                // consumes loads issued last step
        cvt_afrag(A1, n0, n1, n2, n3);    // waits kt+1 A loads; WMMAs in flight

        // issue loads for step kt+2 into buffer 0
        loadB(B0, wbase + (size_t)(kt + 2) * KT_STRIDE);
        const float* p2 = xrow + (kt + 2) * 32;
        v4f m0 = *(const v4f*)(p2);
        v4f m1 = *(const v4f*)(p2 + 4);
        v4f m2 = *(const v4f*)(p2 + 8);
        v4f m3 = *(const v4f*)(p2 + 12);

        mma8(acc, A1, B1);
        cvt_afrag(A0, m0, m1, m2, m3);
    }

    // ---- epilogue: kt = 30, 31 ----
    {
        loadB(B1, wbase + (size_t)(K_TILES - 1) * KT_STRIDE);
        const float* p1 = xrow + (K_TILES - 1) * 32;
        v4f n0 = *(const v4f*)(p1);
        v4f n1 = *(const v4f*)(p1 + 4);
        v4f n2 = *(const v4f*)(p1 + 8);
        v4f n3 = *(const v4f*)(p1 + 12);

        mma8(acc, A0, B0);
        cvt_afrag(A1, n0, n1, n2, n3);
        mma8(acc, A1, B1);
    }

    // C layout: VGPR v -> row = v + 8*(lane>>4), col = lane&15
    const int colb = ngrp * 128 + (lane & 15);
    const int rowb = mtile * 16 + h * 8;
#pragma unroll
    for (int g = 0; g < 8; ++g)
#pragma unroll
        for (int v = 0; v < 8; ++v)
            feat[(size_t)(rowb + v) * NHC + colb + g * 16] = acc[g][v];
}

// ---------------------------------------------------------------------------
// K2: el[n,h] = dot(feat[n,h,:], attn_l[h,:]);  er likewise. 1 thread / (n,h).
// ---------------------------------------------------------------------------
__global__ __launch_bounds__(256) void gat_logits(const float* __restrict__ feat,
                                                  const float* __restrict__ attn_l,
                                                  const float* __restrict__ attn_r,
                                                  float* __restrict__ el,
                                                  float* __restrict__ er) {
    int tid = blockIdx.x * 256 + threadIdx.x;
    if (tid >= N_NODES * HEADS) return;
    int n = tid >> 3, h = tid & 7;
    const float* f  = feat + (size_t)n * NHC + h * OUT_CH;
    const float* al = attn_l + h * OUT_CH;
    const float* ar = attn_r + h * OUT_CH;
    float sl = 0.f, sr = 0.f;
#pragma unroll
    for (int c4 = 0; c4 < OUT_CH / 4; ++c4) {
        v4f fv = *(const v4f*)(f + c4 * 4);
        v4f lv = *(const v4f*)(al + c4 * 4);
        v4f rv = *(const v4f*)(ar + c4 * 4);
#pragma unroll
        for (int j = 0; j < 4; ++j) {
            sl += fv[j] * lv[j];
            sr += fv[j] * rv[j];
        }
    }
    el[tid] = sl;
    er[tid] = sr;
}

// ---------------------------------------------------------------------------
// K3: init out=0, emax=-inf, denom=0.
// ---------------------------------------------------------------------------
__global__ __launch_bounds__(256) void gat_init(float* __restrict__ out,
                                                float* __restrict__ emax,
                                                float* __restrict__ denom) {
    int tid = blockIdx.x * 256 + threadIdx.x;
    if (tid < N_NODES * NHC) out[tid] = 0.f;
    if (tid < N_NODES * HEADS) {
        emax[tid]  = __uint_as_float(0xFF800000u);   // -inf
        denom[tid] = 0.f;
    }
}

static __device__ __forceinline__ void atomicMaxF(float* addr, float val) {
    if (val >= 0.f) atomicMax((int*)addr, __float_as_int(val));
    else            atomicMin((unsigned int*)addr, __float_as_uint(val));
}

// ---------------------------------------------------------------------------
// K4: segment max of leaky_relu(el[src]+er[dst]) over dst. 1 thread / (e,h).
// ---------------------------------------------------------------------------
__global__ __launch_bounds__(256) void gat_edge_max(const float* __restrict__ el,
                                                    const float* __restrict__ er,
                                                    const int* __restrict__ src,
                                                    const int* __restrict__ dst,
                                                    float* __restrict__ emax) {
    int tid = blockIdx.x * 256 + threadIdx.x;
    if (tid >= N_EDGES * HEADS) return;
    int e = tid >> 3, h = tid & 7;
    int s = src[e], d = dst[e];
    float v = el[s * HEADS + h] + er[d * HEADS + h];
    v = v > 0.f ? v : 0.2f * v;
    atomicMaxF(&emax[d * HEADS + h], v);
}

// ---------------------------------------------------------------------------
// K5: ex = exp(e - emax[dst]); denom[dst] += ex (atomic); store ex per edge.
// ---------------------------------------------------------------------------
__global__ __launch_bounds__(256) void gat_edge_exp(const float* __restrict__ el,
                                                    const float* __restrict__ er,
                                                    const int* __restrict__ src,
                                                    const int* __restrict__ dst,
                                                    const float* __restrict__ emax,
                                                    float* __restrict__ exbuf,
                                                    float* __restrict__ denom) {
    int tid = blockIdx.x * 256 + threadIdx.x;
    if (tid >= N_EDGES * HEADS) return;
    int e = tid >> 3, h = tid & 7;
    int s = src[e], d = dst[e];
    float v = el[s * HEADS + h] + er[d * HEADS + h];
    v = v > 0.f ? v : 0.2f * v;
    float ex = __expf(v - emax[d * HEADS + h]);
    exbuf[tid] = ex;
    atomicAdd(&denom[d * HEADS + h], ex);
}

// ---------------------------------------------------------------------------
// K6: out[dst] += feat[src] * a   (a = ex/denom), one wave per edge.
// Lane l covers channel l of head i (i = 0..7) -> 128 B coalesced atomics.
// feat & out fit in the 192 MB L2, so this is L2-atomic bound.
// ---------------------------------------------------------------------------
__global__ __launch_bounds__(256) void gat_scatter(const float* __restrict__ feat,
                                                   const int* __restrict__ src,
                                                   const int* __restrict__ dst,
                                                   const float* __restrict__ exbuf,
                                                   const float* __restrict__ denom,
                                                   float* __restrict__ out) {
    int w    = threadIdx.x >> 5;
    int lane = threadIdx.x & 31;
    int e    = blockIdx.x * 8 + w;
    if (e >= N_EDGES) return;                        // uniform per wave
    int sn = src[e], dn = dst[e];
    const float* fs  = feat + (size_t)sn * NHC;
    float*       op  = out  + (size_t)dn * NHC;
    const float* exb = exbuf + (size_t)e * HEADS;
    const float* dnb = denom + (size_t)dn * HEADS;
#pragma unroll
    for (int i = 0; i < HEADS; ++i) {
        float a = exb[i] / fmaxf(dnb[i], 1e-16f);
        atomicAdd(op + i * 32 + lane, fs[i * 32 + lane] * a);
    }
}

// ---------------------------------------------------------------------------
// Host launcher
// ---------------------------------------------------------------------------
extern "C" void kernel_launch(void* const* d_in, const int* in_sizes, int n_in,
                              void* d_out, int out_size, void* d_ws, size_t ws_size,
                              hipStream_t stream) {
    const float* x      = (const float*)d_in[0];   // [50000,1024]
    const float* W      = (const float*)d_in[1];   // [1024,256]
    const float* attn_l = (const float*)d_in[2];   // [8,32]
    const float* attn_r = (const float*)d_in[3];   // [8,32]
    const int*   src    = (const int*)d_in[4];     // [800000]
    const int*   dst    = (const int*)d_in[5];     // [800000]
    float* out = (float*)d_out;                    // [50000,8,32]

    // workspace carve-up (bytes)
    char* ws = (char*)d_ws;
    float*          feat  = (float*)(ws);                                   // 51,200,000 B
    unsigned short* Wpack = (unsigned short*)(ws + 51200000);               //    524,288 B
    float*          el    = (float*)(ws + 51724288);                        //  1,600,000 B
    float*          er    = (float*)(ws + 53324288);                        //  1,600,000 B
    float*          emax  = (float*)(ws + 54924288);                        //  1,600,000 B
    float*          denom = (float*)(ws + 56524288);                        //  1,600,000 B
    float*          exbuf = (float*)(ws + 58124288);                        // 25,600,000 B
    (void)ws_size; (void)in_sizes; (void)n_in; (void)out_size;

    gat_pack_w <<<64,    256, 0, stream>>>(W, Wpack);
    gat_init   <<<50000, 256, 0, stream>>>(out, emax, denom);
    gat_gemm   <<<(M_TILES + 3) / 4, 256, 0, stream>>>(x, Wpack, feat);     // 782 blocks
    gat_logits <<<(N_NODES * HEADS + 255) / 256, 256, 0, stream>>>(feat, attn_l, attn_r, el, er);
    gat_edge_max<<<(N_EDGES * HEADS) / 256, 256, 0, stream>>>(el, er, src, dst, emax);
    gat_edge_exp<<<(N_EDGES * HEADS) / 256, 256, 0, stream>>>(el, er, src, dst, emax, exbuf, denom);
    gat_scatter<<<N_EDGES / 8, 256, 0, stream>>>(feat, src, dst, exbuf, denom, out);
}
